// Cluster_56985626083513
// MI455X (gfx1250) — matrix operand order repaired and checked by reading, not compiled, not run on hardware
//
#include <hip/hip_runtime.h>
#include <math.h>

typedef __bf16 bf16_t;
typedef bf16_t v16bf __attribute__((ext_vector_type(16)));
typedef bf16_t v8bf  __attribute__((ext_vector_type(8)));
typedef bf16_t v4bf  __attribute__((ext_vector_type(4)));
typedef float  v8f   __attribute__((ext_vector_type(8)));
typedef float  v4f   __attribute__((ext_vector_type(4)));
typedef unsigned int uint32x4 __attribute__((ext_vector_type(4)));
typedef int          int32x8  __attribute__((ext_vector_type(8)));
typedef int          int32x4  __attribute__((ext_vector_type(4)));

#define DIM_ 256
#define S_   4096   // 64*64 spatial per (b, channel)

// ---------------------------------------------------------------------------
// TDM: 1-row tile of `ndwords` dwords, contiguous global -> contiguous LDS.
// D# layout per CDNA5 ISA (group0 128b, group1 256b).
// Issue from one wave only; completion via TENSORcnt.
// 6-arg builtin form (this toolchain): (g0, g1, g2, g3, g4, cpol).
// ---------------------------------------------------------------------------
__device__ __forceinline__ void tdm_load_1d(void* lds_ptr, const void* gptr,
                                            unsigned ndwords) {
  const unsigned long long ga = (unsigned long long)gptr;
  const unsigned lds_off = (unsigned)(unsigned long long)lds_ptr;  // addr[31:0] = LDS offset
  uint32x4 g0;
  g0[0] = 1u;                                            // count=1, load, user mode
  g0[1] = lds_off;                                       // lds_addr (bytes)
  g0[2] = (unsigned)(ga & 0xFFFFFFFFu);                  // global_addr[31:0]
  g0[3] = (unsigned)((ga >> 32) & 0x01FFFFFFu)           // global_addr[56:32]
        | (2u << 30);                                    // type=2 ("image")
  int32x8 g1;
  g1[0] = (int)(2u << 16);                               // data_size=4B
  g1[1] = (int)((ndwords & 0xFFFFu) << 16);              // tensor_dim0[15:0]
  g1[2] = (int)((ndwords >> 16) & 0xFFFFu)               // tensor_dim0[31:16]
        | (int)(1u << 16);                               // tensor_dim1 = 1
  g1[3] = (int)((ndwords & 0xFFFFu) << 16);              // tile_dim0 = ndwords
  g1[4] = 1;                                             // tile_dim1 = 1, tile_dim2 = 0
  g1[5] = (int)ndwords;                                  // tensor_dim0_stride[31:0]
  g1[6] = 0;
  g1[7] = 0;
  int32x4 gz;
  gz[0] = 0; gz[1] = 0; gz[2] = 0; gz[3] = 0;
  int32x8 gz8;
  #pragma unroll
  for (int i = 0; i < 8; ++i) gz8[i] = 0;
  __builtin_amdgcn_tensor_load_to_lds(g0, g1, gz, gz, gz8, 0);
}

// ---------------------------------------------------------------------------
// GEMM: out[o,s] = sum_k W[o,k] * x[b,k,s] + bias[o]
// Dual mode: virtual O=512; rows [0,256) -> (w0,b0,out0), [256,512) -> (w1,b1,out1).
// Block tile: 64(o) x 128(s), 8 waves, each wave 32x32 via 2x2 WMMA accumulators.
// ---------------------------------------------------------------------------
__global__ __launch_bounds__(256) void conv1x1_wmma(
    const float* __restrict__ x,
    const float* __restrict__ w0, const float* __restrict__ b0, float* __restrict__ out0,
    const float* __restrict__ w1, const float* __restrict__ b1, float* __restrict__ out1)
{
  __shared__ bf16_t As[64 * 32];    // [o][k]
  __shared__ bf16_t Bs[128 * 32];   // [s][k]  (x tile transposed)

  const int tid  = threadIdx.x;
  const int lane = tid & 31;
  const int wave = tid >> 5;
  const int b      = blockIdx.z;
  const int s_base = blockIdx.x * 128;
  int o_base = blockIdx.y * 64;

  const float* W; const float* bias; float* out;
  if (o_base < 256) { W = w0; bias = b0; out = out0; }
  else              { W = w1; bias = b1; out = out1; o_base -= 256; }

  const int wo  = wave >> 2;     // 0..1 -> o offset 32*wo
  const int ws  = wave & 3;      // 0..3 -> s offset 32*ws
  const int n16 = lane & 15;
  const int hi  = lane >> 4;

  // staging decomposition: A -> 8 consecutive k per thread; B -> 4k x 4s block
  const int a_row = tid >> 2, a_col = (tid & 3) * 8;
  const int kg = tid >> 5;          // 8 groups of 4 k-rows
  const int sg = tid & 31;          // 32 groups of 4 s
  const int k0 = kg * 4, s0 = sg * 4;

  v8f acc[2][2];
  #pragma unroll
  for (int i = 0; i < 2; ++i)
    #pragma unroll
    for (int j = 0; j < 2; ++j)
      #pragma unroll
      for (int r = 0; r < 8; ++r) acc[i][j][r] = 0.0f;

  for (int kt = 0; kt < DIM_ / 32; ++kt) {
    // ---- stage A: 64x32 weight tile, packed bf16, one b128 store ----
    {
      const float* src = W + (size_t)(o_base + a_row) * DIM_ + kt * 32 + a_col;
      v8bf av;
      #pragma unroll
      for (int j = 0; j < 8; ++j) av[j] = (bf16_t)src[j];
      *(v8bf*)(As + a_row * 32 + a_col) = av;
    }
    // ---- stage B: 4(k)x4(s) block per thread; b64 stores, k contiguous ----
    {
      v4f xr[4];
      #pragma unroll
      for (int r = 0; r < 4; ++r)
        xr[r] = *(const v4f*)(x + ((size_t)b * DIM_ + kt * 32 + k0 + r) * S_
                                + s_base + s0);
      #pragma unroll
      for (int j = 0; j < 4; ++j) {
        v4bf bv;
        #pragma unroll
        for (int r = 0; r < 4; ++r) bv[r] = (bf16_t)xr[r][j];
        *(v4bf*)(Bs + (s0 + j) * 32 + k0) = bv;
      }
      if (kt + 1 < DIM_ / 32)
        __builtin_prefetch(x + ((size_t)b * DIM_ + (kt + 1) * 32 + k0) * S_
                             + s_base + s0, 0, 1);
    }
    __syncthreads();

    // ---- fragments per documented 16-bit A(16x32)/B(32x16) layouts ----
    v16bf afrag[2], bfrag[2];
    #pragma unroll
    for (int fo = 0; fo < 2; ++fo) {
      const int o_l = wo * 32 + fo * 16 + n16;                // M = lane%16
      union { v16bf v; v8bf h[2]; } u;
      u.h[0] = *(const v8bf*)(As + o_l * 32 + hi * 8);        // K = hi*8 .. +8
      u.h[1] = *(const v8bf*)(As + o_l * 32 + 16 + hi * 8);   // K = 16+hi*8 .. +8
      afrag[fo] = u.v;
    }
    #pragma unroll
    for (int fs = 0; fs < 2; ++fs) {
      const int s_l = ws * 32 + fs * 16 + n16;                // N = lane%16
      union { v16bf v; v8bf h[2]; } u;
      u.h[0] = *(const v8bf*)(Bs + s_l * 32 + hi * 8);        // K = hi*8 .. +8
      u.h[1] = *(const v8bf*)(Bs + s_l * 32 + 16 + hi * 8);   // K = 16+hi*8 .. +8
      bfrag[fs] = u.v;
    }
    #pragma unroll
    for (int fo = 0; fo < 2; ++fo)
      #pragma unroll
      for (int fs = 0; fs < 2; ++fs)
        acc[fo][fs] = __builtin_amdgcn_wmma_f32_16x16x32_bf16(
            false, afrag[fo], false, bfrag[fs], (short)0, acc[fo][fs], false, false);
    __syncthreads();
  }

  // ---- epilogue: C layout lane L: N=L%16, VGPR r -> M = r + 8*(L/16) ----
  #pragma unroll
  for (int fo = 0; fo < 2; ++fo)
    #pragma unroll
    for (int fs = 0; fs < 2; ++fs) {
      const int s = s_base + ws * 32 + fs * 16 + n16;
      #pragma unroll
      for (int r = 0; r < 8; ++r) {
        const int o = o_base + wo * 32 + fo * 16 + r + 8 * hi;
        out[((size_t)b * 256 + o) * S_ + s] = acc[fo][fs][r] + bias[o];
      }
    }
}

// ---------------------------------------------------------------------------
// Pool 4x4 blocks -> centers; L2-normalize feat-centers across c (wave reduce).
// One wave per (b2, m); lane = channel.
// ---------------------------------------------------------------------------
__global__ __launch_bounds__(256) void centers_kernel(
    const float* __restrict__ feat, const float* __restrict__ value,
    float* __restrict__ cent_n, float* __restrict__ vcent)
{
  const int g = blockIdx.x * 8 + (threadIdx.x >> 5);  // [0, 512*64)
  const int c = threadIdx.x & 31;
  const int m = g & 63, b2 = g >> 6;
  const int pw = m >> 3, ph = m & 7;
  const int fh = b2 & 1, fw = (b2 >> 1) & 1, head = (b2 >> 2) & 7, Bi = b2 >> 5;

  const size_t chan = (size_t)Bi * 256 + head * 32 + c;
  const float* fptr = feat  + chan * S_;
  const float* vptr = value + chan * S_;

  float fs = 0.0f, vs = 0.0f;
  #pragma unroll
  for (int dw = 0; dw < 4; ++dw)
    #pragma unroll
    for (int dh = 0; dh < 4; ++dh) {
      const int w = fw * 32 + pw * 4 + dw;
      const int h = fh * 32 + ph * 4 + dh;
      const int off = w * 64 + h;
      fs += fptr[off];
      vs += vptr[off];
    }
  fs *= (1.0f / 16.0f);
  vs *= (1.0f / 16.0f);

  float ssq = fs * fs;
  #pragma unroll
  for (int k = 16; k >= 1; k >>= 1) ssq += __shfl_xor(ssq, k, 32);
  const float fn = fs / fmaxf(sqrtf(ssq), 1e-12f);

  const size_t oidx = ((size_t)b2 * 64 + m) * 32 + c;
  cent_n[oidx] = fn;
  vcent[oidx]  = vs;
}

// ---------------------------------------------------------------------------
// Per-b2 clustering: sims + top-4 + sparse aggregate (LDS atomics) + dispatch.
// Centers / value-centers brought in by TDM (tensor_load_to_lds); the vcent
// DMA overlaps all of pass 1 and is only waited before the out_c phase.
// ---------------------------------------------------------------------------
__global__ __launch_bounds__(256) void cluster_kernel(
    const float* __restrict__ feat, const float* __restrict__ value,
    const float* __restrict__ cent_n, const float* __restrict__ vcent,
    const float* __restrict__ alpha_p, const float* __restrict__ beta_p,
    const float* __restrict__ bis1_p, const float* __restrict__ bis2_p,
    const float* __restrict__ bis3_p,
    float* __restrict__ y)
{
  __shared__ float cent[64 * 32];       // normalized centers; reused as out_c
  __shared__ float vcl[64 * 32];        // value-centers (TDM, overlapped)
  __shared__ float accv[64 * 32];       // weighted value accumulation
  __shared__ float rowsum[64];
  __shared__ float topw[1024 * 4];      // weighted top-4 values per token
  __shared__ unsigned int topm[1024];   // packed 4x8-bit center indices

  const int tid = threadIdx.x;
  const int b2  = blockIdx.x;
  const int fh = b2 & 1, fw = (b2 >> 1) & 1, head = (b2 >> 2) & 7, Bi = b2 >> 5;

  const float alpha = alpha_p[0], beta = beta_p[0];
  const float c1 = bis1_p[0], c2 = bis2_p[0], c3 = bis3_p[0];

  // TDM DMA: wave 0 issues both loads; waits cent (cnt<=1) before pass 1.
  if (tid < 32) {
    tdm_load_1d(cent, cent_n + (size_t)b2 * 2048, 2048);
    tdm_load_1d(vcl,  vcent  + (size_t)b2 * 2048, 2048);
    __builtin_amdgcn_s_wait_tensorcnt((short)1);
  }
  for (int i = tid; i < 64 * 32; i += 256) accv[i] = 0.0f;
  if (tid < 64) rowsum[tid] = 0.0f;
  __syncthreads();

  const size_t cbase = ((size_t)Bi * 256 + head * 32) * (size_t)S_;

  // ---- pass 1: sims, top-4, scatter-add into centers ----
  #pragma unroll 1
  for (int it = 0; it < 4; ++it) {
    const int n  = it * 256 + tid;
    const int w2 = n >> 5, h2 = n & 31;
    const int off = (fw * 32 + w2) * 64 + (fh * 32 + h2);

    float tv[32], vv[32];
    #pragma unroll
    for (int c = 0; c < 32; ++c) tv[c] = feat [cbase + (size_t)c * S_ + off];
    #pragma unroll
    for (int c = 0; c < 32; ++c) vv[c] = value[cbase + (size_t)c * S_ + off];

    float ssq = 0.0f;
    #pragma unroll
    for (int c = 0; c < 32; ++c) ssq += tv[c] * tv[c];
    const float inv = 1.0f / fmaxf(sqrtf(ssq), 1e-12f);

    float v0 = -1e30f, v1 = -1e30f, v2 = -1e30f, v3 = -1e30f;
    int   i0 = 0, i1 = 0, i2 = 0, i3 = 0;
    #pragma unroll 4
    for (int m = 0; m < 64; ++m) {
      float d = 0.0f;
      #pragma unroll
      for (int q = 0; q < 8; ++q) {
        const v4f cv = *(const v4f*)(cent + m * 32 + q * 4);
        d += cv[0] * tv[q * 4 + 0] + cv[1] * tv[q * 4 + 1]
           + cv[2] * tv[q * 4 + 2] + cv[3] * tv[q * 4 + 3];
      }
      const float s = 1.0f / (1.0f + expf(-(beta + alpha * d * inv)));
      if (s > v0)      { v3=v2; i3=i2; v2=v1; i2=i1; v1=v0; i1=i0; v0=s; i0=m; }
      else if (s > v1) { v3=v2; i3=i2; v2=v1; i2=i1; v1=s; i1=m; }
      else if (s > v2) { v3=v2; i3=i2; v2=s; i2=m; }
      else if (s > v3) { v3=s; i3=m; }
    }
    const float w0 = v0, w1 = v1 * c1, w2c = v2 * c2, w3 = v3 * c3;
    topw[n * 4 + 0] = w0; topw[n * 4 + 1] = w1;
    topw[n * 4 + 2] = w2c; topw[n * 4 + 3] = w3;
    topm[n] = (unsigned)i0 | ((unsigned)i1 << 8) | ((unsigned)i2 << 16) | ((unsigned)i3 << 24);

    const float wj4[4] = { w0, w1, w2c, w3 };
    const int   mj4[4] = { i0, i1, i2, i3 };
    #pragma unroll
    for (int j = 0; j < 4; ++j) {
      const float wj = wj4[j]; const int mj = mj4[j];
      atomicAdd(&rowsum[mj], wj);
      #pragma unroll
      for (int c = 0; c < 32; ++c) atomicAdd(&accv[mj * 32 + c], wj * vv[c]);
    }
  }
  if (tid < 32) __builtin_amdgcn_s_wait_tensorcnt((short)0);  // vcl ready
  __syncthreads();

  // ---- out_c = (acc + value_centers) / (rowsum + 1), reuse cent ----
  for (int i = tid; i < 64 * 32; i += 256) {
    const int m = i >> 5;
    cent[i] = (accv[i] + vcl[i]) / (rowsum[m] + 1.0f);
  }
  __syncthreads();

  // ---- pass 2: dispatch tokens from their 4 centers; unfold on store ----
  #pragma unroll 1
  for (int it = 0; it < 4; ++it) {
    const int n  = it * 256 + tid;
    const int w2 = n >> 5, h2 = n & 31;
    const int off = (fw * 32 + w2) * 64 + (fh * 32 + h2);

    const unsigned pm = topm[n];
    float o[32];
    #pragma unroll
    for (int c = 0; c < 32; ++c) o[c] = 0.0f;
    #pragma unroll
    for (int j = 0; j < 4; ++j) {
      const float wj = topw[n * 4 + j];
      const int   mj = (pm >> (8 * j)) & 0xFF;
      #pragma unroll
      for (int q = 0; q < 8; ++q) {
        const v4f cv = *(const v4f*)(cent + mj * 32 + q * 4);
        o[q * 4 + 0] += wj * cv[0]; o[q * 4 + 1] += wj * cv[1];
        o[q * 4 + 2] += wj * cv[2]; o[q * 4 + 3] += wj * cv[3];
      }
    }
    #pragma unroll
    for (int c = 0; c < 32; ++c) y[cbase + (size_t)c * S_ + off] = o[c];
  }
}

// ---------------------------------------------------------------------------
extern "C" void kernel_launch(void* const* d_in, const int* in_sizes, int n_in,
                              void* d_out, int out_size, void* d_ws, size_t ws_size,
                              hipStream_t stream) {
  (void)in_sizes; (void)n_in; (void)out_size; (void)ws_size;
  const float* x     = (const float*)d_in[0];
  const float* f_w   = (const float*)d_in[1];
  const float* f_b   = (const float*)d_in[2];
  const float* v_w   = (const float*)d_in[3];
  const float* v_b   = (const float*)d_in[4];
  const float* p_w   = (const float*)d_in[5];
  const float* p_b   = (const float*)d_in[6];
  const float* alpha = (const float*)d_in[7];
  const float* beta  = (const float*)d_in[8];
  const float* bis1  = (const float*)d_in[9];
  const float* bis2  = (const float*)d_in[10];
  const float* bis3  = (const float*)d_in[11];

  char* ws = (char*)d_ws;
  const size_t SZ = (size_t)16 * 256 * 4096 * sizeof(float);  // 64 MB
  float* feat  = (float*)(ws);
  float* value = (float*)(ws + SZ);
  float* y     = (float*)(ws + 2 * SZ);
  float* cent  = (float*)(ws + 3 * SZ);
  float* vcent = (float*)(ws + 3 * SZ + (size_t)512 * 64 * 32 * sizeof(float));

  // feat + value convs fused (O=512 virtual), shared x tiles
  conv1x1_wmma<<<dim3(32, 8, 16), 256, 0, stream>>>(x, f_w, f_b, feat, v_w, v_b, value);
  // pooled centers + L2 norm
  centers_kernel<<<4096, 256, 0, stream>>>(feat, value, cent, vcent);
  // clustering (top-4 sparse aggregate + dispatch), TDM center loads
  cluster_kernel<<<512, 256, 0, stream>>>(feat, value, cent, vcent,
                                          alpha, beta, bis1, bis2, bis3, y);
  // projection conv
  conv1x1_wmma<<<dim3(32, 4, 16), 256, 0, stream>>>(y, p_w, p_b, (float*)d_out,
                                                    p_w, p_b, (float*)d_out);
}
